// ConditionalVelocity_77489799954843
// MI455X (gfx1250) — matrix operand order repaired and checked by reading, not compiled, not run on hardware
//
#include <hip/hip_runtime.h>

#define HD 128
#define NN 256
#define BBATCH 2
#define TPB 256

typedef __attribute__((ext_vector_type(16))) _Float16 v16h;
typedef __attribute__((ext_vector_type(8)))  float    v8f;

union V16HU { v16h v; uint4 q[2]; };

// ---------------- LDS layout (bytes) ----------------
// [0, 131328) is a contiguous image async-copied straight from global:
//   We1 swizzled | We2 swizzled | Wc1 swizzled | We1 d2-row
#define OFF_WE1   0        // 256x128 f16, WMMA-B swizzled (65536 B)
#define OFF_WE2   65536    // 128x128 f16 swizzled (32768 B)
#define OFF_WC1   98304    // 128x128 f16 swizzled (32768 B)
#define OFF_WE1R  131072   // We1 row 256 (d2 row), 128 f16 (256 B)
#define WIMG_BYTES 131328  // size of the weight image
#define OFF_HJ    131328   // 256x128 f16 row-major (65536 B), async from h16
#define OFF_BUFM  196864   // per-wave 16x128 f16 tiles (8 waves, 32768 B)
#define OFF_BUFC  229632   // per-wave 16x128 f16 tiles (32768 B)
#define OFF_HI    262400   // h_i f16 (256 B)
#define OFF_HIF   262656   // h_i f32 (512 B)
#define OFF_DIFF  263168   // diff[256][3] f32 (3072 B)
#define OFF_D2    266240   // d2[256] f32 (1024 B)
#define OFF_HIB   267264   // h_i@We1 + be1, f32 (512 B)
#define OFF_B2    267776   // be2 f32 (512 B)
#define OFF_BC1   268288   // bc1 f32 (512 B)
#define OFF_WC2   268800   // Wc2 f32 (512 B)
#define OFF_AGG   269312   // agg f32 (512 B)
#define OFF_U     269824   // node-MLP hidden f32 (512 B)
#define OFF_XS    270336   // coord sums (16 B)
#define OFF_SCAL  270352   // bc2 scalar (16 B)
#define SMEM_BYTES 270368

#define PERL 65664         // halves per layer weight image

// Map W[k][n] (row-major, N=128) into the per-lane contiguous WMMA B layout:
// B-tile (kt,nt): lane l holds 16 halves h with K = kt*32 + (l>>4)*16 + h,
// n = nt*16 + (l&15). Tile base = ((kt*8+nt)*32 + l)*16.
__device__ __forceinline__ int swz_idx(int k, int n) {
  int kt = k >> 5, lhi = (k >> 4) & 1, h = k & 15;
  int nt = n >> 4, ln = (lhi << 4) | (n & 15);
  return ((((kt << 3) | nt) << 5) | ln) * 16 + h;
}

__device__ __forceinline__ float silu_f(float x) { return x / (1.0f + __expf(-x)); }

// LDS byte offset of a generic pointer into dynamic shared memory
__device__ __forceinline__ unsigned lds_off(const void* p) {
  return (unsigned)(unsigned long long)(__attribute__((address_space(3))) const char*)p;
}

// Async DMA: 16 bytes global -> LDS, tracked by ASYNCcnt (no VGPR round-trip)
__device__ __forceinline__ void async_b128(unsigned lds, unsigned off,
                                           const void* gbase) {
  asm volatile("global_load_async_to_lds_b128 %0, %1, %2"
               :: "v"(lds), "v"(off), "s"(gbase) : "memory");
}

// 16x128 * 128x128 accumulate: A row-major [16][rowStride] f16 in LDS
// (rowStride==0 => broadcast row), B pre-swizzled in LDS. acc = 8 n-tiles.
__device__ __forceinline__ void mm4(const _Float16* Arows, int rowStride,
                                    const _Float16* Bsw, int lane, v8f acc[8]) {
  const int lhi = lane >> 4, lm = lane & 15;
  #pragma unroll
  for (int kt = 0; kt < 4; ++kt) {
    V16HU A;
    // A layout: lane holds row m=lm; halves 0..15 at K=kt*32+lhi*16+h
    const _Float16* ap = Arows + lm * rowStride + kt * 32 + lhi * 16;
    A.q[0] = *(const uint4*)(ap);
    A.q[1] = *(const uint4*)(ap + 8);
    #pragma unroll
    for (int nt = 0; nt < 8; ++nt) {
      V16HU Bv;
      const _Float16* bp = Bsw + ((kt * 8 + nt) * 32 + lane) * 16;
      Bv.q[0] = *(const uint4*)(bp);
      Bv.q[1] = *(const uint4*)(bp + 8);
      acc[nt] = __builtin_amdgcn_wmma_f32_16x16x32_f16(
          false, A.v, false, Bv.v, (short)0, acc[nt], false, false);
    }
  }
}

// One-time prep: convert+swizzle edge-MLP weights to per-layer f16 LDS images,
// convert h to f16.
__global__ void prep_kernel(const float* __restrict__ h,
                            const float* We1a, const float* We2a, const float* Wc1a,
                            const float* We1b, const float* We2b, const float* Wc1b,
                            const float* We1c, const float* We2c, const float* Wc1c,
                            _Float16* __restrict__ img0, _Float16* __restrict__ img1,
                            _Float16* __restrict__ img2, _Float16* __restrict__ h16) {
  const int total = 3 * PERL + BBATCH * NN * HD;
  for (int idx = blockIdx.x * blockDim.x + threadIdx.x; idx < total;
       idx += gridDim.x * blockDim.x) {
    if (idx < 3 * PERL) {
      const int l = idx / PERL, r = idx - l * PERL;
      const float* We1 = (l == 0) ? We1a : (l == 1) ? We1b : We1c;
      const float* We2 = (l == 0) ? We2a : (l == 1) ? We2b : We2c;
      const float* Wc1 = (l == 0) ? Wc1a : (l == 1) ? Wc1b : Wc1c;
      _Float16* img = (l == 0) ? img0 : (l == 1) ? img1 : img2;
      if (r < 32768) {                       // We1 rows 0..255
        int k = r >> 7, n = r & (HD - 1);
        img[swz_idx(k, n)] = (_Float16)We1[r];
      } else if (r < 49152) {                // We2
        int rr = r - 32768, k = rr >> 7, n = rr & (HD - 1);
        img[32768 + swz_idx(k, n)] = (_Float16)We2[rr];
      } else if (r < 65536) {                // Wc1
        int rr = r - 49152, k = rr >> 7, n = rr & (HD - 1);
        img[49152 + swz_idx(k, n)] = (_Float16)Wc1[rr];
      } else {                               // We1 d2 row (row 256)
        int n = r - 65536;
        img[65536 + n] = (_Float16)We1[256 * HD + n];
      }
    } else {
      int r = idx - 3 * PERL;
      h16[r] = (_Float16)h[r];
    }
  }
}

template <int F_OUT>
__launch_bounds__(TPB, 1)
__global__ void egnn_layer_kernel(
    const float* __restrict__ h_in, const _Float16* __restrict__ h16_in,
    const float* __restrict__ x_in, const _Float16* __restrict__ img,
    const float* __restrict__ be1, const float* __restrict__ be2,
    const float* __restrict__ bc1, const float* __restrict__ Wc2,
    const float* __restrict__ bc2, const float* __restrict__ Wn1,
    const float* __restrict__ bn1, const float* __restrict__ Wn2,
    const float* __restrict__ bn2, float* __restrict__ h_out,
    _Float16* __restrict__ h16_out, float* __restrict__ x_out) {
  extern __shared__ char smem[];
  _Float16* sWe1  = (_Float16*)(smem + OFF_WE1);
  _Float16* sWe2  = (_Float16*)(smem + OFF_WE2);
  _Float16* sWc1  = (_Float16*)(smem + OFF_WC1);
  _Float16* sWe1r = (_Float16*)(smem + OFF_WE1R);
  _Float16* sHj   = (_Float16*)(smem + OFF_HJ);
  _Float16* sBufM = (_Float16*)(smem + OFF_BUFM);
  _Float16* sBufC = (_Float16*)(smem + OFF_BUFC);
  _Float16* sHi   = (_Float16*)(smem + OFF_HI);
  float* sHiF  = (float*)(smem + OFF_HIF);
  float* sDiff = (float*)(smem + OFF_DIFF);
  float* sD2   = (float*)(smem + OFF_D2);
  float* sHiB  = (float*)(smem + OFF_HIB);
  float* sB2   = (float*)(smem + OFF_B2);
  float* sBc1  = (float*)(smem + OFF_BC1);
  float* sWc2  = (float*)(smem + OFF_WC2);
  float* sAgg  = (float*)(smem + OFF_AGG);
  float* sU    = (float*)(smem + OFF_U);
  float* sXs   = (float*)(smem + OFF_XS);
  float* sScal = (float*)(smem + OFF_SCAL);

  const int tid  = threadIdx.x;
  const int lane = tid & 31;
  const int wave = tid >> 5;
  const int lhi  = lane >> 4, lm = lane & 15;
  const int bi = blockIdx.x;
  const int b = bi / NN, i = bi - b * NN;

  // ---- async DMA staging: weight image + f16 h_j straight into LDS ----
  const unsigned ldsBase = lds_off(smem);
  for (unsigned o = tid * 16u; o < WIMG_BYTES; o += TPB * 16u)
    async_b128(ldsBase + o, o, img);
  const void* hjsrc = (const void*)(h16_in + (size_t)b * NN * HD);
  for (unsigned o = tid * 16u; o < NN * HD * 2u; o += TPB * 16u)
    async_b128(ldsBase + OFF_HJ + o, o, hjsrc);

  // ---- small f32 staging in parallel with the DMA ----
  if (tid < HD) {
    float hv = h_in[((size_t)b * NN + i) * HD + tid];
    sHi[tid]  = (_Float16)hv;
    sHiF[tid] = hv;
    sB2[tid] = be2[tid]; sBc1[tid] = bc1[tid];
    sWc2[tid] = Wc2[tid];
    sAgg[tid] = 0.0f;
  }
  const float xi0 = x_in[(b * NN + i) * 3 + 0];
  const float xi1 = x_in[(b * NN + i) * 3 + 1];
  const float xi2 = x_in[(b * NN + i) * 3 + 2];
  if (tid < NN) {
    float dx = xi0 - x_in[(b * NN + tid) * 3 + 0];
    float dy = xi1 - x_in[(b * NN + tid) * 3 + 1];
    float dz = xi2 - x_in[(b * NN + tid) * 3 + 2];
    sDiff[tid * 3 + 0] = dx; sDiff[tid * 3 + 1] = dy; sDiff[tid * 3 + 2] = dz;
    sD2[tid] = dx * dx + dy * dy + dz * dz;
  }
  if (tid < 4) sXs[tid] = 0.0f;
  if (tid == 0) sScal[0] = bc2[0];
  asm volatile("s_wait_asynccnt 0" ::: "memory");
  __syncthreads();

  // ---- once per workgroup: hiB = h_i @ We1[0:128] + be1 (row-constant half
  // of e_in). Each wave computes one 16-wide n-tile with 4 broadcast-A WMMAs;
  // row M=0 of the C layout lives in lanes 0..15, element 0.
  {
    v8f acch;
    #pragma unroll
    for (int r = 0; r < 8; ++r) acch[r] = 0.0f;
    const int nt = wave;
    #pragma unroll
    for (int kt = 0; kt < 4; ++kt) {
      V16HU A;
      const _Float16* ap = sHi + kt * 32 + lhi * 16;  // broadcast row
      A.q[0] = *(const uint4*)(ap);
      A.q[1] = *(const uint4*)(ap + 8);
      V16HU Bv;
      const _Float16* bp = sWe1 + ((kt * 8 + nt) * 32 + lane) * 16;
      Bv.q[0] = *(const uint4*)(bp);
      Bv.q[1] = *(const uint4*)(bp + 8);
      acch = __builtin_amdgcn_wmma_f32_16x16x32_f16(
          false, A.v, false, Bv.v, (short)0, acch, false, false);
    }
    if (lane < 16) {
      const int n = nt * 16 + lane;
      sHiB[n] = acch[0] + be1[n];
    }
  }
  __syncthreads();

  _Float16* bufM = sBufM + wave * (16 * HD);
  _Float16* bufC = sBufC + wave * (16 * HD);

  // each wave: 2 tiles of 16 neighbors j, fully fused edge-MLP chain
  for (int t = 0; t < 2; ++t) {
    const int j0 = (wave * 2 + t) * 16;
    v8f acc[8];

    // GEMM1 (h_j half only): m1 = silu(hj@We1_hj + hiB + d2*We1_row)
    #pragma unroll
    for (int nt = 0; nt < 8; ++nt)
      #pragma unroll
      for (int r = 0; r < 8; ++r) acc[nt][r] = 0.0f;
    mm4(sHj + j0 * HD, HD, sWe1 + 4 * 8 * 32 * 16, lane, acc);  // h_j half
    float d2v[8];
    #pragma unroll
    for (int r = 0; r < 8; ++r) d2v[r] = sD2[j0 + r + 8 * lhi];
    #pragma unroll
    for (int nt = 0; nt < 8; ++nt) {
      const int n = nt * 16 + lm;
      const float wr = (float)sWe1r[n];
      const float bv = sHiB[n];                     // h_i contribution + be1
      #pragma unroll
      for (int r = 0; r < 8; ++r) {                 // C layout: m=r+8*lhi, n=lm
        float v = acc[nt][r] + d2v[r] * wr + bv;    // rank-1 d2 column in f32
        bufM[(r + 8 * lhi) * HD + n] = (_Float16)silu_f(v);
      }
    }
    asm volatile("s_wait_dscnt 0" ::: "memory");

    // GEMM2: m = silu(m1 @ We2 + be2);  agg += sum_j m
    #pragma unroll
    for (int nt = 0; nt < 8; ++nt)
      #pragma unroll
      for (int r = 0; r < 8; ++r) acc[nt][r] = 0.0f;
    mm4(bufM, HD, sWe2, lane, acc);
    #pragma unroll
    for (int nt = 0; nt < 8; ++nt) {
      const int n = nt * 16 + lm;
      const float bv = sB2[n];
      float part = 0.0f;
      #pragma unroll
      for (int r = 0; r < 8; ++r) {
        float v = silu_f(acc[nt][r] + bv);
        part += v;
        bufC[(r + 8 * lhi) * HD + n] = (_Float16)v;
      }
      atomicAdd(&sAgg[n], part);                    // ds_add_f32
    }
    asm volatile("s_wait_dscnt 0" ::: "memory");

    // GEMM3: c1 = silu(m @ Wc1 + bc1)
    #pragma unroll
    for (int nt = 0; nt < 8; ++nt)
      #pragma unroll
      for (int r = 0; r < 8; ++r) acc[nt][r] = 0.0f;
    mm4(bufC, HD, sWc1, lane, acc);
    #pragma unroll
    for (int nt = 0; nt < 8; ++nt) {
      const int n = nt * 16 + lm;
      const float bv = sBc1[n];
      #pragma unroll
      for (int r = 0; r < 8; ++r)
        bufM[(r + 8 * lhi) * HD + n] = (_Float16)silu_f(acc[nt][r] + bv);
    }
    asm volatile("s_wait_dscnt 0" ::: "memory");

    // GEMV: w_j = c1 . Wc2 + bc2 ; xsum += diff_j * w_j
    float wp = 0.0f;
    const int f0 = lhi * 64;
    #pragma unroll 8
    for (int f = 0; f < 64; ++f)
      wp += (float)bufM[lm * HD + f0 + f] * sWc2[f0 + f];
    wp += __shfl_down(wp, 16, 32);
    if (lane < 16) {
      const float wj = wp + sScal[0];
      const int j = j0 + lane;
      atomicAdd(&sXs[0], sDiff[j * 3 + 0] * wj);
      atomicAdd(&sXs[1], sDiff[j * 3 + 1] * wj);
      atomicAdd(&sXs[2], sDiff[j * 3 + 2] * wj);
    }
  }
  __syncthreads();

  // coordinate update: x_new = x + mean_j(diff * w)
  if (tid < 3)
    x_out[(b * NN + i) * 3 + tid] =
        x_in[(b * NN + i) * 3 + tid] + sXs[tid] * (1.0f / NN);

  // node MLP: h_new = tanh(silu([h_i | agg] @ Wn1 + bn1) @ Wn2 + bn2)
  if (tid < HD) {
    float a = bn1[tid];
    #pragma unroll 4
    for (int k = 0; k < HD; ++k) a += sHiF[k] * Wn1[k * HD + tid];
    #pragma unroll 4
    for (int k = 0; k < HD; ++k) a += sAgg[k] * Wn1[(HD + k) * HD + tid];
    sU[tid] = silu_f(a);
  }
  __syncthreads();
  if (tid < F_OUT) {
    float a = bn2[tid];
    #pragma unroll 4
    for (int k = 0; k < HD; ++k) a += sU[k] * Wn2[k * F_OUT + tid];
    float hv = tanhf(a);
    h_out[(b * NN + i) * F_OUT + tid] = hv;
    if (F_OUT == HD) h16_out[((size_t)b * NN + i) * HD + tid] = (_Float16)hv;
  }
}

// v = noise * exp(log_sigma); v -= mean over nodes
__global__ void velocity_kernel(const float* __restrict__ noise,
                                const float* __restrict__ ls,
                                float* __restrict__ out) {
  __shared__ float ssum[4];
  const int b = blockIdx.x, n = threadIdx.x;
  if (n < 4) ssum[n] = 0.0f;
  __syncthreads();
  const float e = __expf(ls[b * NN + n]);
  float v0 = noise[(b * NN + n) * 3 + 0] * e;
  float v1 = noise[(b * NN + n) * 3 + 1] * e;
  float v2 = noise[(b * NN + n) * 3 + 2] * e;
  atomicAdd(&ssum[0], v0);
  atomicAdd(&ssum[1], v1);
  atomicAdd(&ssum[2], v2);
  __syncthreads();
  out[(b * NN + n) * 3 + 0] = v0 - ssum[0] * (1.0f / NN);
  out[(b * NN + n) * 3 + 1] = v1 - ssum[1] * (1.0f / NN);
  out[(b * NN + n) * 3 + 2] = v2 - ssum[2] * (1.0f / NN);
}

extern "C" void kernel_launch(void* const* d_in, const int* in_sizes, int n_in,
                              void* d_out, int out_size, void* d_ws, size_t ws_size,
                              hipStream_t stream) {
  (void)in_sizes; (void)n_in; (void)out_size; (void)ws_size;
  const float* h     = (const float*)d_in[0];
  const float* x     = (const float*)d_in[1];
  const float* noise = (const float*)d_in[2];
  const float* P[3][12];
  for (int l = 0; l < 3; ++l)
    for (int a = 0; a < 12; ++a)
      P[l][a] = (const float*)d_in[3 + l * 12 + a];
  // per-layer param order: We1 be1 We2 be2 Wc1 bc1 Wc2 bc2 Wn1 bn1 Wn2 bn2

  const int BN = BBATCH * NN;
  float* hA = (float*)d_ws;
  float* hB = hA + BN * HD;
  float* xA = hB + BN * HD;
  float* xB = xA + BN * 3;
  float* ls = xB + BN * 3;  // [B,N] log_sigma
  float* xC = ls + BN;
  _Float16* h16_0 = (_Float16*)(xC + BN * 3);
  _Float16* h16_1 = h16_0 + BN * HD;
  _Float16* h16_2 = h16_1 + BN * HD;
  _Float16* img0  = h16_2 + BN * HD;
  _Float16* img1  = img0 + PERL;
  _Float16* img2  = img1 + PERL;

  prep_kernel<<<dim3(128), dim3(TPB), 0, stream>>>(
      h, P[0][0], P[0][2], P[0][4], P[1][0], P[1][2], P[1][4],
      P[2][0], P[2][2], P[2][4], img0, img1, img2, h16_0);

  dim3 grid(BN), block(TPB);
  egnn_layer_kernel<HD><<<grid, block, SMEM_BYTES, stream>>>(
      h, h16_0, x, img0, P[0][1], P[0][3], P[0][5], P[0][6], P[0][7],
      P[0][8], P[0][9], P[0][10], P[0][11], hA, h16_1, xA);
  egnn_layer_kernel<HD><<<grid, block, SMEM_BYTES, stream>>>(
      hA, h16_1, xA, img1, P[1][1], P[1][3], P[1][5], P[1][6], P[1][7],
      P[1][8], P[1][9], P[1][10], P[1][11], hB, h16_2, xB);
  egnn_layer_kernel<1><<<grid, block, SMEM_BYTES, stream>>>(
      hB, h16_2, xB, img2, P[2][1], P[2][3], P[2][5], P[2][6], P[2][7],
      P[2][8], P[2][9], P[2][10], P[2][11], ls, h16_2, xC);
  velocity_kernel<<<dim3(BBATCH), dim3(NN), 0, stream>>>(noise, ls,
                                                         (float*)d_out);
}
